// Euler_Attention_11184094839239
// MI455X (gfx1250) — compile-verified, matrix operand order checked by reading, not run on hardware
//
#include <hip/hip_runtime.h>
#include <hip/hip_bf16.h>
#include <math.h>

// ---------------- problem constants ----------------
namespace {
constexpr int B = 2, L = 2048, D = 1024, H = 16, Dh = 64;
constexpr int M = B * L;        // 4096 rows for full-batch GEMMs
constexpr int Dhalf = D / 2;    // 512
}

typedef unsigned short u16;
typedef unsigned int   u32;
typedef unsigned long long u64;
typedef __bf16 bf16x16 __attribute__((ext_vector_type(16)));
typedef float  f32x8   __attribute__((ext_vector_type(8)));

struct alignas(16) U4v { u32 x, y, z, w; };
union FragBF { bf16x16 v; U4v u4[2]; u32 u[8]; };

__device__ __forceinline__ u32 f2bf(float f) {
  u32 u = __float_as_uint(f);
  return (u + 0x7FFFu + ((u >> 16) & 1u)) >> 16;   // round-to-nearest-even
}
__device__ __forceinline__ u16 f2bf16(float f) { return (u16)f2bf(f); }
__device__ __forceinline__ u32 pack2(float a, float b) { return f2bf(a) | (f2bf(b) << 16); }

// ---------------- f32 -> bf16 conversion ----------------
__global__ void f32_to_bf16_kernel(const float* __restrict__ src, u16* __restrict__ dst, int n) {
  int t = blockIdx.x * blockDim.x + threadIdx.x;
  if (t < n) dst[t] = f2bf16(src[t]);
}

// ---------------- bf16 WMMA GEMM, 64x64 wave tile: C[m,n] = sum_k X[m,k]*W[n,k] + bias[n] ----
// X: Mrows x K bf16 row-major, W: N x K bf16 row-major, C: Mrows x N f32.
// Block = 8 waves in a 2(M) x 4(N) grid -> block tile 128 x 256.
// Batched via blockIdx.z with element strides bsX/bsW/bsC.
__global__ __launch_bounds__(256) void wmma_gemm_kernel(
    const u16* __restrict__ X, const u16* __restrict__ W,
    const float* __restrict__ bias, float* __restrict__ C,
    int Mrows, int N, int K, long bsX, long bsW, long bsC)
{
  int z = blockIdx.z;
  X += (size_t)bsX * z; W += (size_t)bsW * z; C += (size_t)bsC * z;
  int lane = threadIdx.x & 31;
  int wave = threadIdx.x >> 5;
  int wm = wave >> 2, wn = wave & 3;
  int tm = blockIdx.y * 128 + wm * 64;
  int tn = blockIdx.x * 256 + wn * 64;
  if (tm >= Mrows) return;
  int r = lane & 15, h = lane >> 4;
  const u16* Ar0 = X + (size_t)(tm + r) * K;   // A row m = lane&15 (+i*16)
  const u16* Br0 = W + (size_t)(tn + r) * K;   // B col n = lane&15 (+j*16), reads row n of W
  f32x8 acc[4][4];
  #pragma unroll
  for (int i = 0; i < 4; ++i)
    #pragma unroll
    for (int j = 0; j < 4; ++j)
      acc[i][j] = f32x8{0.f,0.f,0.f,0.f,0.f,0.f,0.f,0.f};

  #pragma unroll 1
  for (int k0 = 0; k0 < K; k0 += 32) {
    FragBF a[4], bfr[4];
    #pragma unroll
    for (int i = 0; i < 4; ++i) {
      const u16* Ar = Ar0 + (size_t)i * 16 * K + k0;
      a[i].u4[0] = *(const U4v*)(Ar + h * 8);        // K = k0 + half*8 .. +7
      a[i].u4[1] = *(const U4v*)(Ar + 16 + h * 8);   // K = k0+16+half*8 .. +7
      const u16* Br = Br0 + (size_t)i * 16 * K + k0;
      bfr[i].u4[0] = *(const U4v*)(Br + h * 8);
      bfr[i].u4[1] = *(const U4v*)(Br + 16 + h * 8);
    }
    #pragma unroll
    for (int i = 0; i < 4; ++i)
      #pragma unroll
      for (int j = 0; j < 4; ++j)
        acc[i][j] = __builtin_amdgcn_wmma_f32_16x16x32_bf16(
            false, a[i].v, false, bfr[j].v, (short)0, acc[i][j], false, false);
  }

  #pragma unroll
  for (int j = 0; j < 4; ++j) {
    int cn = tn + j * 16 + r;
    float bv = bias ? bias[cn] : 0.0f;
    #pragma unroll
    for (int i = 0; i < 4; ++i) {
      float* Cp = C + (size_t)(tm + i * 16 + h * 8) * N + cn;  // vgpr e -> row +e
      #pragma unroll
      for (int e = 0; e < 8; ++e) Cp[(size_t)e * N] = acc[i][j][e] + bv;
    }
  }
}

// ---------------- scores = mean over L ----------------
__global__ void mean_l_kernel(const float* __restrict__ q, float* __restrict__ s) {
  int t = blockIdx.x * blockDim.x + threadIdx.x;
  if (t >= B * D) return;
  int b = t / D, d = t % D;
  const float* p = q + (size_t)b * L * D + d;
  float acc = 0.f;
  for (int l = 0; l < L; ++l) acc += p[(size_t)l * D];
  s[t] = acc * (1.0f / (float)L);
}

// ---------------- Bsum[j] = sum_k |s_j - s_k| ----------------
__global__ void bsum_kernel(const float* __restrict__ s, float* __restrict__ bs) {
  int t = blockIdx.x * blockDim.x + threadIdx.x;
  if (t >= B * D) return;
  int b = t / D;
  float sj = s[t];
  const float* sb = s + (size_t)b * D;
  float acc = 0.f;
  for (int k = 0; k < D; ++k) acc += fabsf(sj - sb[k]);
  bs[t] = acc;
}

// ---------------- P[i,j] = softmax_j(scaling_i * s_j - Bsum_j), bf16 out ----------------
__global__ __launch_bounds__(256) void neural_sort_p_kernel(
    const float* __restrict__ s, const float* __restrict__ bs, u16* __restrict__ P)
{
  int b = blockIdx.x / D, i = blockIdx.x % D;
  const float* sb = s + (size_t)b * D;
  const float* bb = bs + (size_t)b * D;
  float scal = (float)(D - 1 - 2 * i);       // n + 1 - 2*(i+1)
  float v[4], ex[4];
  float mx = -3.0e38f;
  #pragma unroll
  for (int e = 0; e < 4; ++e) {
    int j = threadIdx.x + e * 256;
    v[e] = scal * sb[j] - bb[j];
    mx = fmaxf(mx, v[e]);
  }
  __shared__ float red[256];
  red[threadIdx.x] = mx; __syncthreads();
  for (int st = 128; st > 0; st >>= 1) {
    if (threadIdx.x < st) red[threadIdx.x] = fmaxf(red[threadIdx.x], red[threadIdx.x + st]);
    __syncthreads();
  }
  mx = red[0]; __syncthreads();
  float ls = 0.f;
  #pragma unroll
  for (int e = 0; e < 4; ++e) { ex[e] = __expf(v[e] - mx); ls += ex[e]; }
  red[threadIdx.x] = ls; __syncthreads();
  for (int st = 128; st > 0; st >>= 1) {
    if (threadIdx.x < st) red[threadIdx.x] += red[threadIdx.x + st];
    __syncthreads();
  }
  float inv = 1.0f / red[0];
  u16* Pr = P + ((size_t)b * D + i) * D;
  #pragma unroll
  for (int e = 0; e < 4; ++e) {
    int j = threadIdx.x + e * 256;
    Pr[j] = f2bf16(ex[e] * inv);
  }
}

// ---------------- Euler transform (elementwise), bf16 out ----------------
__global__ void euler_kernel(const float* __restrict__ xs,
                             const float* __restrict__ delta,
                             const float* __restrict__ beul,
                             const float* __restrict__ lgs,
                             u16* __restrict__ out, int isq)
{
  int t = blockIdx.x * blockDim.x + threadIdx.x;
  if (t >= B * L * Dhalf) return;
  int c = t % Dhalf;
  size_t base = (size_t)(t / Dhalf) * D + 2 * c;
  float r = xs[base], p = xs[base + 1];
  float lam = sqrtf(r * r + p * p + 1e-6f);
  float th = atan2f(p, r) * delta[c] + (isq ? beul[c] : 0.0f);
  lam *= __expf(fminf(fmaxf(lgs[c], -5.0f), 5.0f));
  float sn, cs;
  __sincosf(th, &sn, &cs);
  out[base]     = f2bf16(lam * cs);
  out[base + 1] = f2bf16(lam * sn);
}

// ---------------- V transpose to [b,h,d,L] bf16 ----------------
__global__ void vtranspose_kernel(const float* __restrict__ v, u16* __restrict__ vt) {
  int t = blockIdx.x * blockDim.x + threadIdx.x;
  if (t >= B * L * D) return;
  int l = t % L;
  int rest = t / L;
  int d = rest % Dh;
  int bh = rest / Dh;             // b*H + h
  int b = bh / H, h = bh % H;
  vt[t] = f2bf16(v[((size_t)b * L + l) * D + h * Dh + d]);
}

// ---------------- flash attention with async global->LDS K/V staging ----------------
// Block = 8 waves x 16 queries of one (b,h). 32 keys per iteration.
// K tile (32 keys x 64 d) and V^T tile (64 d x 32 keys) are staged once per block
// per iteration via GLOBAL_LOAD_ASYNC_TO_LDS_B128 (ASYNCcnt) and read as WMMA
// fragments by all 8 waves via ds_load_b128 -> 8x less L2 traffic for K/V.
// S^T = K x Q^T -> softmax per D-layout column; S^T D-layout maps directly onto
// the B-operand layout of O^T = V^T x P (zero cross-lane movement).
__global__ __launch_bounds__(256) void flash_attn_kernel(
    const u16* __restrict__ Q, const u16* __restrict__ Kc,
    const u16* __restrict__ Vt, float* __restrict__ ctx)
{
  __shared__ u16 ktile[32 * 64];     // [key][d]
  __shared__ u16 vtile[64 * 32];     // [d][key]

  constexpr int QB = 128;            // 8 waves * 16 queries
  int nq = L / QB;
  int qblk = blockIdx.x % nq;
  int bh = blockIdx.x / nq;
  int b = bh / H, h = bh % H;
  int t = threadIdx.x;
  int wave = t >> 5, lane = t & 31;
  int r = lane & 15, hf = lane >> 4;
  int qbase = qblk * QB + wave * 16;

  // Q fragments (B operand, col n = query), loaded once
  const u16* qrow = Q + ((size_t)b * L + qbase + r) * D + h * Dh;
  FragBF bq0, bq1;
  bq0.u4[0] = *(const U4v*)(qrow + hf * 8);
  bq0.u4[1] = *(const U4v*)(qrow + 16 + hf * 8);
  bq1.u4[0] = *(const U4v*)(qrow + 32 + hf * 8);
  bq1.u4[1] = *(const U4v*)(qrow + 48 + hf * 8);

  // per-thread staging source pointers + LDS byte addresses (16B per thread)
  const u16* kgp = Kc + ((size_t)b * L + (t >> 3)) * D + h * Dh + (t & 7) * 8;
  const u16* vgp = Vt + ((size_t)bh * Dh + (t >> 2)) * L + (t & 3) * 8;
  u32 klds = (u32)(uintptr_t)ktile + (u32)t * 16u;
  u32 vlds = (u32)(uintptr_t)vtile + (u32)t * 16u;

  f32x8 o0 = {0.f,0.f,0.f,0.f,0.f,0.f,0.f,0.f};
  f32x8 o1 = o0, o2 = o0, o3 = o0;
  float mrow = -3.0e38f, lrow = 0.f;
  const float scale = 0.125f;        // 1/sqrt(64)

  #pragma unroll 1
  for (int k0 = 0; k0 < L; k0 += 32) {
    __syncthreads();                 // previous iteration's LDS reads complete
    u64 ka = (u64)(uintptr_t)(kgp + (size_t)k0 * D);
    u64 va = (u64)(uintptr_t)(vgp + k0);
    asm volatile("global_load_async_to_lds_b128 %0, %1, off"
                 :: "v"(klds), "v"(ka) : "memory");
    asm volatile("global_load_async_to_lds_b128 %0, %1, off"
                 :: "v"(vlds), "v"(va) : "memory");
    asm volatile("s_wait_asynccnt 0x0" ::: "memory");
    __syncthreads();                 // all waves' async loads landed in LDS

    // S^T tiles from LDS: A = K rows (key = lane&15 [+16]), B = Q frags
    f32x8 s0 = {0.f,0.f,0.f,0.f,0.f,0.f,0.f,0.f};
    f32x8 s1 = s0;
    FragBF a;
    const u16* kt0 = ktile + r * 64;
    const u16* kt1 = ktile + (16 + r) * 64;
    a.u4[0] = *(const U4v*)(kt0 + hf * 8);
    a.u4[1] = *(const U4v*)(kt0 + 16 + hf * 8);
    s0 = __builtin_amdgcn_wmma_f32_16x16x32_bf16(false, a.v, false, bq0.v, (short)0, s0, false, false);
    a.u4[0] = *(const U4v*)(kt0 + 32 + hf * 8);
    a.u4[1] = *(const U4v*)(kt0 + 48 + hf * 8);
    s0 = __builtin_amdgcn_wmma_f32_16x16x32_bf16(false, a.v, false, bq1.v, (short)0, s0, false, false);
    a.u4[0] = *(const U4v*)(kt1 + hf * 8);
    a.u4[1] = *(const U4v*)(kt1 + 16 + hf * 8);
    s1 = __builtin_amdgcn_wmma_f32_16x16x32_bf16(false, a.v, false, bq0.v, (short)0, s1, false, false);
    a.u4[0] = *(const U4v*)(kt1 + 32 + hf * 8);
    a.u4[1] = *(const U4v*)(kt1 + 48 + hf * 8);
    s1 = __builtin_amdgcn_wmma_f32_16x16x32_bf16(false, a.v, false, bq1.v, (short)0, s1, false, false);

    // online softmax per query column (all 8 regs of a lane belong to col lane&15)
    float ml = -3.0e38f;
    #pragma unroll
    for (int i = 0; i < 8; ++i) {
      s0[i] *= scale; s1[i] *= scale;
      ml = fmaxf(ml, fmaxf(s0[i], s1[i]));
    }
    ml = fmaxf(ml, __shfl_xor(ml, 16, 32));
    float mnew = fmaxf(mrow, ml);
    float corr = __expf(mrow - mnew);
    float p0[8], p1[8], lsl = 0.f;
    #pragma unroll
    for (int i = 0; i < 8; ++i) {
      p0[i] = __expf(s0[i] - mnew);
      p1[i] = __expf(s1[i] - mnew);
      lsl += p0[i] + p1[i];
    }
    lsl += __shfl_xor(lsl, 16, 32);
    lrow = lrow * corr + lsl;
    mrow = mnew;
    #pragma unroll
    for (int i = 0; i < 8; ++i) { o0[i] *= corr; o1[i] *= corr; o2[i] *= corr; o3[i] *= corr; }

    // pack P^T (D layout) straight into the B-operand layout for K=32
    FragBF pb;
    #pragma unroll
    for (int j = 0; j < 4; ++j) {
      pb.u[j]     = pack2(p0[2 * j], p0[2 * j + 1]);
      pb.u[4 + j] = pack2(p1[2 * j], p1[2 * j + 1]);
    }
    // O^T += V^T x P  (4 d-tiles of 16) -- V^T fragments from LDS
    FragBF av;
    const u16* vt0 = vtile + r * 32;
    av.u4[0] = *(const U4v*)(vt0 + hf * 8);
    av.u4[1] = *(const U4v*)(vt0 + 16 + hf * 8);
    o0 = __builtin_amdgcn_wmma_f32_16x16x32_bf16(false, av.v, false, pb.v, (short)0, o0, false, false);
    vt0 += 16 * 32;
    av.u4[0] = *(const U4v*)(vt0 + hf * 8);
    av.u4[1] = *(const U4v*)(vt0 + 16 + hf * 8);
    o1 = __builtin_amdgcn_wmma_f32_16x16x32_bf16(false, av.v, false, pb.v, (short)0, o1, false, false);
    vt0 += 16 * 32;
    av.u4[0] = *(const U4v*)(vt0 + hf * 8);
    av.u4[1] = *(const U4v*)(vt0 + 16 + hf * 8);
    o2 = __builtin_amdgcn_wmma_f32_16x16x32_bf16(false, av.v, false, pb.v, (short)0, o2, false, false);
    vt0 += 16 * 32;
    av.u4[0] = *(const U4v*)(vt0 + hf * 8);
    av.u4[1] = *(const U4v*)(vt0 + 16 + hf * 8);
    o3 = __builtin_amdgcn_wmma_f32_16x16x32_bf16(false, av.v, false, pb.v, (short)0, o3, false, false);
  }

  float inv = 1.0f / lrow;
  int qq = qbase + r;                        // lane's query column
  float* cp = ctx + ((size_t)b * L + qq) * D + h * Dh;
  #pragma unroll
  for (int i = 0; i < 8; ++i) {
    int d = i + 8 * hf;
    cp[d]      = o0[i] * inv;
    cp[16 + d] = o1[i] * inv;
    cp[32 + d] = o2[i] * inv;
    cp[48 + d] = o3[i] * inv;
  }
}

// ---------------- residual + LayerNorm ----------------
__global__ __launch_bounds__(256) void ln_kernel(
    const float* __restrict__ hid, const float* __restrict__ x,
    const float* __restrict__ gamma, const float* __restrict__ beta,
    float* __restrict__ out)
{
  int row = blockIdx.x;
  const float* hp = hid + (size_t)row * D;
  const float* xp = x + (size_t)row * D;
  float v[4]; float s = 0.f;
  #pragma unroll
  for (int e = 0; e < 4; ++e) {
    int d = threadIdx.x + e * 256;
    v[e] = hp[d] + xp[d];
    s += v[e];
  }
  __shared__ float red[256];
  red[threadIdx.x] = s; __syncthreads();
  for (int st = 128; st > 0; st >>= 1) {
    if (threadIdx.x < st) red[threadIdx.x] += red[threadIdx.x + st];
    __syncthreads();
  }
  float mu = red[0] * (1.0f / (float)D);
  __syncthreads();
  float vs = 0.f;
  #pragma unroll
  for (int e = 0; e < 4; ++e) { float tt = v[e] - mu; vs += tt * tt; }
  red[threadIdx.x] = vs; __syncthreads();
  for (int st = 128; st > 0; st >>= 1) {
    if (threadIdx.x < st) red[threadIdx.x] += red[threadIdx.x + st];
    __syncthreads();
  }
  float var = red[0] * (1.0f / (float)D);
  float inv = rsqrtf(var + 1e-12f);
  #pragma unroll
  for (int e = 0; e < 4; ++e) {
    int d = threadIdx.x + e * 256;
    out[(size_t)row * D + d] = (v[e] - mu) * inv * gamma[d] + beta[d];
  }
}

// ---------------- launcher ----------------
extern "C" void kernel_launch(void* const* d_in, const int* in_sizes, int n_in,
                              void* d_out, int out_size, void* d_ws, size_t ws_size,
                              hipStream_t stream)
{
  const float* x     = (const float*)d_in[0];
  const float* Wq    = (const float*)d_in[1];
  const float* bq    = (const float*)d_in[2];
  const float* Wk    = (const float*)d_in[3];
  const float* bk    = (const float*)d_in[4];
  const float* Wv    = (const float*)d_in[5];
  const float* bv    = (const float*)d_in[6];
  const float* Wd    = (const float*)d_in[7];
  const float* bd    = (const float*)d_in[8];
  const float* gamma = (const float*)d_in[9];
  const float* beta  = (const float*)d_in[10];
  const float* delta = (const float*)d_in[11];
  const float* beul  = (const float*)d_in[12];
  const float* lgs   = (const float*)d_in[13];
  float* out = (float*)d_out;
  (void)in_sizes; (void)n_in; (void)out_size; (void)ws_size;

  char* ws = (char*)d_ws;
  size_t o = 0;
  auto al = [&](size_t n) { size_t r = o; o += (n + 255) & ~(size_t)255; return r; };
  const size_t NE = (size_t)B * L * D;     // 4M activation elems
  const size_t WE = (size_t)D * D;         // 1M weight elems
  size_t oE1 = al(NE * 2);                 // xbf -> q_ebf -> ctx_bf
  size_t oW  = al(4 * WE * 2);             // Wq/Wk/Wv/Wd bf16
  size_t oQ  = al(NE * 4);                 // q f32 -> q_sorted -> hidden
  size_t oK  = al(NE * 4);                 // k f32 -> k_sorted
  size_t oV  = al(NE * 4);                 // v f32 -> ctx
  size_t oA  = al(NE * 2);                 // q_bf -> k_bf -> k_ebf
  size_t oVt = al(NE * 2);                 // V^T bf16 [b,h,d,L]
  size_t oP  = al((size_t)B * WE * 2);     // P bf16 (reused for q and k paths)
  size_t oSq = al((size_t)B * D * 4);
  size_t oSk = al((size_t)B * D * 4);
  size_t oBq = al((size_t)B * D * 4);
  size_t oBk = al((size_t)B * D * 4);

  u16* xbf  = (u16*)(ws + oE1);
  u16* wqb  = (u16*)(ws + oW);
  u16* wkb  = wqb + WE;
  u16* wvb  = wqb + 2 * WE;
  u16* wdb  = wqb + 3 * WE;
  float* qf = (float*)(ws + oQ);
  float* kf = (float*)(ws + oK);
  float* vf = (float*)(ws + oV);
  u16* bufA = (u16*)(ws + oA);
  u16* vt   = (u16*)(ws + oVt);
  u16* Pm   = (u16*)(ws + oP);
  float* sq  = (float*)(ws + oSq);
  float* sk  = (float*)(ws + oSk);
  float* bsq = (float*)(ws + oBq);
  float* bsk = (float*)(ws + oBk);

  dim3 blk(256);
  int gN = (int)((NE + 255) / 256);
  int gW = (int)((WE + 255) / 256);
  int gBD = (B * D + 255) / 256;
  int gEul = (B * L * Dhalf + 255) / 256;
  dim3 ggemm(D / 256, M / 128, 1);          // full-batch GEMM: 4096 x 1024 x 1024
  dim3 ggemm2(D / 256, L / 128, B);         // per-batch sort-apply GEMM

  // 1) bf16 copies of x and weights
  f32_to_bf16_kernel<<<gN, blk, 0, stream>>>(x, xbf, (int)NE);
  f32_to_bf16_kernel<<<gW, blk, 0, stream>>>(Wq, wqb, (int)WE);
  f32_to_bf16_kernel<<<gW, blk, 0, stream>>>(Wk, wkb, (int)WE);
  f32_to_bf16_kernel<<<gW, blk, 0, stream>>>(Wv, wvb, (int)WE);
  f32_to_bf16_kernel<<<gW, blk, 0, stream>>>(Wd, wdb, (int)WE);

  // 2) q/k/v projections
  wmma_gemm_kernel<<<ggemm, blk, 0, stream>>>(xbf, wqb, bq, qf, M, D, D, 0, 0, 0);
  wmma_gemm_kernel<<<ggemm, blk, 0, stream>>>(xbf, wkb, bk, kf, M, D, D, 0, 0, 0);
  wmma_gemm_kernel<<<ggemm, blk, 0, stream>>>(xbf, wvb, bv, vf, M, D, D, 0, 0, 0);

  // 3) neural-sort statistics
  mean_l_kernel<<<gBD, blk, 0, stream>>>(qf, sq);
  mean_l_kernel<<<gBD, blk, 0, stream>>>(kf, sk);
  bsum_kernel<<<gBD, blk, 0, stream>>>(sq, bsq);
  bsum_kernel<<<gBD, blk, 0, stream>>>(sk, bsk);

  // 4) q path: P, sort-apply GEMM, Euler -> q_ebf (reuses xbf slot)
  neural_sort_p_kernel<<<B * D, blk, 0, stream>>>(sq, bsq, Pm);
  f32_to_bf16_kernel<<<gN, blk, 0, stream>>>(qf, bufA, (int)NE);
  wmma_gemm_kernel<<<ggemm2, blk, 0, stream>>>(bufA, Pm, nullptr, qf, L, D, D,
                                               (long)L * D, (long)D * D, (long)L * D);
  euler_kernel<<<gEul, blk, 0, stream>>>(qf, delta, beul, lgs, (u16*)(ws + oE1), 1);

  // 5) k path: P, sort-apply GEMM, Euler -> k_ebf (reuses bufA)
  neural_sort_p_kernel<<<B * D, blk, 0, stream>>>(sk, bsk, Pm);
  f32_to_bf16_kernel<<<gN, blk, 0, stream>>>(kf, bufA, (int)NE);
  wmma_gemm_kernel<<<ggemm2, blk, 0, stream>>>(bufA, Pm, nullptr, kf, L, D, D,
                                               (long)L * D, (long)D * D, (long)L * D);
  euler_kernel<<<gEul, blk, 0, stream>>>(kf, delta, beul, lgs, bufA, 0);

  // 6) V^T bf16
  vtranspose_kernel<<<gN, blk, 0, stream>>>(vf, vt);

  // 7) flash attention -> ctx (reuses vf)
  flash_attn_kernel<<<B * H * (L / 128), blk, 0, stream>>>(
      (const u16*)(ws + oE1), bufA, vt, vf);

  // 8) output dense: hidden = ctx @ Wd^T + bd (ctx_bf reuses oE1, hidden reuses qf)
  f32_to_bf16_kernel<<<gN, blk, 0, stream>>>(vf, (u16*)(ws + oE1), (int)NE);
  wmma_gemm_kernel<<<ggemm, blk, 0, stream>>>((const u16*)(ws + oE1), wdb, bd, qf,
                                              M, D, D, 0, 0, 0);

  // 9) residual + LayerNorm -> d_out
  ln_kernel<<<B * L, blk, 0, stream>>>(qf, x, gamma, beta, out);
}
